// SelfAttention_19473381720482
// MI455X (gfx1250) — matrix operand order repaired and checked by reading, not compiled, not run on hardware
//
#include <hip/hip_runtime.h>
#include <hip/hip_bf16.h>

// Problem constants (from reference setup_inputs)
#define BB 4
#define CC 512
#define NN 4096
#define CQ 64

typedef __attribute__((ext_vector_type(8)))  float  v8f;
typedef __attribute__((ext_vector_type(8)))  __bf16 v8bf;
typedef __attribute__((ext_vector_type(16))) __bf16 v16bf;

__device__ __forceinline__ v16bf cat8(v8bf lo, v8bf hi) {
  return __builtin_shufflevector(lo, hi, 0,1,2,3,4,5,6,7,8,9,10,11,12,13,14,15);
}

// A-operand fragment (16x32 bf16, M x K): lane m = lane&15,
// elems 0..7  = K[koff .. koff+7], elems 8..15 = K[16+koff .. 23+koff], koff = (lane>=16)*8
__device__ __forceinline__ v16bf frag_a_bf16(const __bf16* rowp, int lane) {
  int koff = (lane >> 4) << 3;
  v8bf lo = *(const v8bf*)(rowp + koff);
  v8bf hi = *(const v8bf*)(rowp + 16 + koff);
  return cat8(lo, hi);
}

// B-operand fragment (32x16 bf16, K x N): lane n = lane&15,
// elems 0..7 = K[koff .. koff+7], elems 8..15 = K[16+koff .. 23+koff], koff = (lane>=16)*8
__device__ __forceinline__ v16bf frag_b_bf16(const __bf16* rowp, int lane) {
  int koff = (lane >> 4) << 3;
  v8bf lo = *(const v8bf*)(rowp + koff);
  v8bf hi = *(const v8bf*)(rowp + 16 + koff);
  return cat8(lo, hi);
}

#define WMMA_BF16(a, b, c) \
  __builtin_amdgcn_wmma_f32_16x16x32_bf16(false, (a), false, (b), (short)0, (c), false, false)

// CDNA5 async copy: global memory -> LDS, 16B per lane, tracked by ASYNCcnt.
__device__ __forceinline__ void async_ld16(unsigned int lds_off, const void* gptr) {
  asm volatile("global_load_async_to_lds_b128 %0, %1, off"
               :: "v"(lds_off), "v"(gptr) : "memory");
}
__device__ __forceinline__ void wait_async0() {
  asm volatile("s_wait_asynccnt 0x0" ::: "memory");
}

// ---------------------------------------------------------------------------
// 0) Elementwise fp32 -> bf16 convert (for weights, done once)
// ---------------------------------------------------------------------------
__global__ __launch_bounds__(256) void cvt_bf16_kernel(const float* __restrict__ src,
                                                       __bf16* __restrict__ dst, int n) {
  int i = blockIdx.x * 256 + threadIdx.x;
  if (i < n) dst[i] = (__bf16)src[i];
}

// ---------------------------------------------------------------------------
// 1) xT[b][n][c] (bf16) = x[b][c][n] (fp32)   — 32x32 LDS transpose tiles
// ---------------------------------------------------------------------------
__global__ __launch_bounds__(256) void transpose_x_kernel(const float* __restrict__ x,
                                                          __bf16* __restrict__ xT) {
  __shared__ float tile[32][33];
  int b = blockIdx.z;
  int n0 = blockIdx.x * 32;
  int c0 = blockIdx.y * 32;
  int tx = threadIdx.x, ty = threadIdx.y; // 32 x 8
#pragma unroll
  for (int k = 0; k < 32; k += 8)
    tile[ty + k][tx] = x[((size_t)b * CC + c0 + ty + k) * NN + n0 + tx];
  __syncthreads();
#pragma unroll
  for (int k = 0; k < 32; k += 8)
    xT[((size_t)b * NN + n0 + ty + k) * CC + c0 + tx] = (__bf16)tile[tx][ty + k];
}

// ---------------------------------------------------------------------------
// 2) outT[b][n][o] = sum_c xT[b][n][c] * W[o][c]   (M=4096 rows, N=64, K=512)
//    Used for fT (W=Wq) and gT (W=Wk). 8 waves x 16 rows = 128 rows/block.
// ---------------------------------------------------------------------------
__global__ __launch_bounds__(256) void proj_small_kernel(const __bf16* __restrict__ xT,
                                                         const __bf16* __restrict__ W,
                                                         __bf16* __restrict__ outT) {
  int b = blockIdx.y;
  int n0 = blockIdx.x * 128;
  int tid = threadIdx.x, lane = tid & 31, wid = tid >> 5;
  int m0 = wid * 16;
  const __bf16* xb = xT + (size_t)b * NN * CC;

  v8f acc[4] = {};
  for (int k0 = 0; k0 < CC; k0 += 32) {
    v16bf a = frag_a_bf16(&xb[(size_t)(n0 + m0 + (lane & 15)) * CC + k0], lane);
#pragma unroll
    for (int t = 0; t < 4; ++t) {
      v16bf bfr = frag_b_bf16(&W[(size_t)(t * 16 + (lane & 15)) * CC + k0], lane);
      acc[t] = WMMA_BF16(a, bfr, acc[t]);
    }
  }
  int hi8 = (lane >> 4) << 3;
#pragma unroll
  for (int t = 0; t < 4; ++t)
#pragma unroll
    for (int e = 0; e < 8; ++e) {
      int n = n0 + m0 + e + hi8;
      int o = t * 16 + (lane & 15);
      outT[((size_t)b * NN + n) * CQ + o] = (__bf16)acc[t][e];
    }
}

// ---------------------------------------------------------------------------
// 3) h[b][co][n] = sum_c Wv[co][c] * xT[b][n][c]   (M=512, N=4096, K=512)
//    128x128 tile / block; 8 waves as 2(M) x 4(N); wave tile 64x32.
// ---------------------------------------------------------------------------
__global__ __launch_bounds__(256) void proj_h_kernel(const __bf16* __restrict__ Wv,
                                                     const __bf16* __restrict__ xT,
                                                     __bf16* __restrict__ h) {
  int b = blockIdx.z;
  int n0 = blockIdx.x * 128;
  int c0 = blockIdx.y * 128;
  int tid = threadIdx.x, lane = tid & 31, wid = tid >> 5;
  int wm = wid & 1, wn = wid >> 1;
  const __bf16* xb = xT + (size_t)b * NN * CC;

  v8f acc[4][2] = {};
  for (int k0 = 0; k0 < CC; k0 += 32) {
    v16bf bfr[2];
#pragma unroll
    for (int t = 0; t < 2; ++t)
      bfr[t] = frag_b_bf16(&xb[(size_t)(n0 + wn * 32 + t * 16 + (lane & 15)) * CC + k0], lane);
#pragma unroll
    for (int r = 0; r < 4; ++r) {
      v16bf a = frag_a_bf16(&Wv[(size_t)(c0 + wm * 64 + r * 16 + (lane & 15)) * CC + k0], lane);
#pragma unroll
      for (int t = 0; t < 2; ++t) acc[r][t] = WMMA_BF16(a, bfr[t], acc[r][t]);
    }
  }
  int hi8 = (lane >> 4) << 3;
#pragma unroll
  for (int r = 0; r < 4; ++r)
#pragma unroll
    for (int t = 0; t < 2; ++t)
#pragma unroll
      for (int e = 0; e < 8; ++e) {
        int c = c0 + wm * 64 + r * 16 + e + hi8;
        int n = n0 + wn * 32 + t * 16 + (lane & 15);
        h[((size_t)b * CC + c) * NN + n] = (__bf16)acc[r][t][e];
      }
}

// ---------------------------------------------------------------------------
// 4) sT[b][j][i] = sum_c gT[b][j][c] * fT[b][i][c]   (M=N=4096, K=64)
// ---------------------------------------------------------------------------
__global__ __launch_bounds__(256) void scores_kernel(const __bf16* __restrict__ gT,
                                                     const __bf16* __restrict__ fT,
                                                     float* __restrict__ sT) {
  int b = blockIdx.z;
  int i0 = blockIdx.x * 128;
  int j0 = blockIdx.y * 128;
  int tid = threadIdx.x, lane = tid & 31, wid = tid >> 5;
  int wm = wid & 1, wn = wid >> 1;
  const __bf16* gb = gT + (size_t)b * NN * CQ;
  const __bf16* fb = fT + (size_t)b * NN * CQ;

  v8f acc[4][2] = {};
#pragma unroll
  for (int k0 = 0; k0 < CQ; k0 += 32) {
    v16bf bfr[2];
#pragma unroll
    for (int t = 0; t < 2; ++t)
      bfr[t] = frag_b_bf16(&fb[(size_t)(i0 + wn * 32 + t * 16 + (lane & 15)) * CQ + k0], lane);
#pragma unroll
    for (int r = 0; r < 4; ++r) {
      v16bf a = frag_a_bf16(&gb[(size_t)(j0 + wm * 64 + r * 16 + (lane & 15)) * CQ + k0], lane);
#pragma unroll
      for (int t = 0; t < 2; ++t) acc[r][t] = WMMA_BF16(a, bfr[t], acc[r][t]);
    }
  }
  int hi8 = (lane >> 4) << 3;
#pragma unroll
  for (int r = 0; r < 4; ++r)
#pragma unroll
    for (int t = 0; t < 2; ++t)
#pragma unroll
      for (int e = 0; e < 8; ++e) {
        int j = j0 + wm * 64 + r * 16 + e + hi8;
        int i = i0 + wn * 32 + t * 16 + (lane & 15);
        sT[((size_t)b * NN + j) * NN + i] = acc[r][t][e];
      }
}

// ---------------------------------------------------------------------------
// 5) Row softmax of sT (axis i is contiguous) -> betaT bf16, same layout
// ---------------------------------------------------------------------------
__global__ __launch_bounds__(256) void softmax_kernel(const float* __restrict__ sT,
                                                      __bf16* __restrict__ bT) {
  int j = blockIdx.x, b = blockIdx.y;
  const float* row = sT + ((size_t)b * NN + j) * NN;
  __bf16* orow = bT + ((size_t)b * NN + j) * NN;
  int t = threadIdx.x, lane = t & 31, wid = t >> 5;

  float v[16];
  float mx = -3.0e38f;
#pragma unroll
  for (int k = 0; k < 16; ++k) { v[k] = row[t + 256 * k]; mx = fmaxf(mx, v[k]); }
#pragma unroll
  for (int o = 16; o > 0; o >>= 1) mx = fmaxf(mx, __shfl_xor(mx, o, 32));
  __shared__ float redm[8];
  if (lane == 0) redm[wid] = mx;
  __syncthreads();
  mx = redm[0];
#pragma unroll
  for (int w = 1; w < 8; ++w) mx = fmaxf(mx, redm[w]);

  float s = 0.f;
#pragma unroll
  for (int k = 0; k < 16; ++k) { v[k] = __expf(v[k] - mx); s += v[k]; }
#pragma unroll
  for (int o = 16; o > 0; o >>= 1) s += __shfl_xor(s, o, 32);
  __shared__ float reds[8];
  if (lane == 0) reds[wid] = s;
  __syncthreads();
  s = 0.f;
#pragma unroll
  for (int w = 0; w < 8; ++w) s += reds[w];
  float inv = 1.0f / s;
#pragma unroll
  for (int k = 0; k < 16; ++k) orow[t + 256 * k] = (__bf16)(v[k] * inv);
}

// ---------------------------------------------------------------------------
// 6) out[b][c][m] = gamma * sum_n h[b][c][n]*betaT[b][m][n] + x[b][c][m]
//    M=512(c), N=4096(m), K=4096(n). Double-buffered ASYNC global->LDS
//    staging (global_load_async_to_lds_b128 + s_wait_asynccnt) overlapped
//    with the WMMA burst on the previous tile.
// ---------------------------------------------------------------------------
#define LPITCH 40
__global__ __launch_bounds__(256) void attn_out_kernel(const __bf16* __restrict__ h,
                                                       const __bf16* __restrict__ bT,
                                                       const float* __restrict__ x,
                                                       const float* __restrict__ gamma,
                                                       float* __restrict__ out) {
  __shared__ __bf16 sm[2][2][128 * LPITCH]; // [buffer][A=h / B=betaT]
  int b = blockIdx.z;
  int m0 = blockIdx.x * 128; // output positions
  int c0 = blockIdx.y * 128; // channels
  int tid = threadIdx.x, lane = tid & 31, wid = tid >> 5;
  int wm = wid & 1, wn = wid >> 1;
  const __bf16* hb = h + (size_t)b * CC * NN;
  const __bf16* pb = bT + (size_t)b * NN * NN;

  // Per-thread staging slots: 2 chunks of 16B for each of A and B per tile.
  int row0 = tid >> 1, ch0 = (tid & 1) << 1;        // chunks {ch0, ch0+1} of row0? no:
  // 512 chunks per tile: idx = tid + i*256, row = idx>>2, ch = idx&3
  v8f acc[4][2] = {};

  // Prologue: stage tile 0 into buffer 0
#pragma unroll
  for (int i = 0; i < 2; ++i) {
    int idx = tid + i * 256, row = idx >> 2, ch = idx & 3;
    async_ld16((unsigned int)(uintptr_t)&sm[0][0][row * LPITCH + ch * 8],
               &hb[(size_t)(c0 + row) * NN + ch * 8]);
    async_ld16((unsigned int)(uintptr_t)&sm[0][1][row * LPITCH + ch * 8],
               &pb[(size_t)(m0 + row) * NN + ch * 8]);
  }

  const int NIT = NN / 32;
  for (int it = 0; it < NIT; ++it) {
    int cur = it & 1;
    wait_async0();      // tile `it` fully landed in LDS (this wave's copies)
    __syncthreads();    // ... and everyone else's; prior readers of other buf done
    if (it + 1 < NIT) {
      int k1 = (it + 1) * 32, nxt = cur ^ 1;
#pragma unroll
      for (int i = 0; i < 2; ++i) {
        int idx = tid + i * 256, row = idx >> 2, ch = idx & 3;
        async_ld16((unsigned int)(uintptr_t)&sm[nxt][0][row * LPITCH + ch * 8],
                   &hb[(size_t)(c0 + row) * NN + k1 + ch * 8]);
        async_ld16((unsigned int)(uintptr_t)&sm[nxt][1][row * LPITCH + ch * 8],
                   &pb[(size_t)(m0 + row) * NN + k1 + ch * 8]);
      }
    }
    v16bf bfr[2];
#pragma unroll
    for (int t = 0; t < 2; ++t)
      bfr[t] = frag_b_bf16(&sm[cur][1][(wn * 32 + t * 16 + (lane & 15)) * LPITCH], lane);
#pragma unroll
    for (int r = 0; r < 4; ++r) {
      v16bf a = frag_a_bf16(&sm[cur][0][(wm * 64 + r * 16 + (lane & 15)) * LPITCH], lane);
#pragma unroll
      for (int t = 0; t < 2; ++t) acc[r][t] = WMMA_BF16(a, bfr[t], acc[r][t]);
    }
  }
  float g = gamma[0];
  int hi8 = (lane >> 4) << 3;
#pragma unroll
  for (int r = 0; r < 4; ++r)
#pragma unroll
    for (int t = 0; t < 2; ++t)
#pragma unroll
      for (int e = 0; e < 8; ++e) {
        int c = c0 + wm * 64 + r * 16 + e + hi8;
        int m = m0 + wn * 32 + t * 16 + (lane & 15);
        size_t off = ((size_t)b * CC + c) * NN + m;
        out[off] = g * acc[r][t][e] + x[off];
      }
}

// ---------------------------------------------------------------------------
extern "C" void kernel_launch(void* const* d_in, const int* in_sizes, int n_in,
                              void* d_out, int out_size, void* d_ws, size_t ws_size,
                              hipStream_t stream) {
  (void)in_sizes; (void)n_in; (void)out_size; (void)ws_size;
  const float* x  = (const float*)d_in[0];
  const float* Wq = (const float*)d_in[1];
  const float* Wk = (const float*)d_in[2];
  const float* Wv = (const float*)d_in[3];
  const float* gm = (const float*)d_in[4];
  float* out = (float*)d_out;

  char* ws = (char*)d_ws;
  const size_t MB = 1ull << 20;
  __bf16* xT = (__bf16*)(ws);                 // 16 MB : [B][N][C]
  __bf16* fT = (__bf16*)(ws + 16 * MB);       //  2 MB : [B][N][CQ]
  __bf16* gT = (__bf16*)(ws + 18 * MB);       //  2 MB : [B][N][CQ]
  __bf16* hB = (__bf16*)(ws + 20 * MB);       // 16 MB : [B][C][N]
  float*  sT = (float*)(ws + 36 * MB);        // 256 MB: [B][N][N]
  __bf16* bT = (__bf16*)(ws + 36 * MB + (size_t)BB * NN * NN * 4); // 128 MB
  __bf16* Wqb = (__bf16*)(ws + 420 * MB);            // 64 KB
  __bf16* Wkb = (__bf16*)(ws + 420 * MB + 65536);    // 64 KB
  __bf16* Wvb = (__bf16*)(ws + 420 * MB + 131072);   // 512 KB

  cvt_bf16_kernel<<<dim3((CQ * CC + 255) / 256), 256, 0, stream>>>(Wq, Wqb, CQ * CC);
  cvt_bf16_kernel<<<dim3((CQ * CC + 255) / 256), 256, 0, stream>>>(Wk, Wkb, CQ * CC);
  cvt_bf16_kernel<<<dim3((CC * CC + 255) / 256), 256, 0, stream>>>(Wv, Wvb, CC * CC);
  transpose_x_kernel<<<dim3(NN / 32, CC / 32, BB), dim3(32, 8), 0, stream>>>(x, xT);
  proj_small_kernel<<<dim3(NN / 128, BB), 256, 0, stream>>>(xT, Wqb, fT);
  proj_small_kernel<<<dim3(NN / 128, BB), 256, 0, stream>>>(xT, Wkb, gT);
  proj_h_kernel<<<dim3(NN / 128, CC / 128, BB), 256, 0, stream>>>(Wvb, xT, hB);
  scores_kernel<<<dim3(NN / 128, NN / 128, BB), 256, 0, stream>>>(gT, fT, sT);
  softmax_kernel<<<dim3(NN, BB), 256, 0, stream>>>(sT, bT);
  attn_out_kernel<<<dim3(NN / 128, CC / 128, BB), 256, 0, stream>>>(hB, bT, x, gm, out);
}
